// SA_fusion_56899726737880
// MI455X (gfx1250) — compile-verified, matrix operand order checked by reading, not run on hardware
//
#include <hip/hip_runtime.h>
#include <hip/hip_bf16.h>

// ---------------------------------------------------------------------------
// Problem constants (fixed by the reference): B=4, C=256, H=W=64 -> HW=4096
// ---------------------------------------------------------------------------
#define BATCH 4
#define CCH   256
#define HWP   4096
#define SPITCH 4104                    // padded LDS pitch (floats), 16B aligned
#define SMEM_FLOATS (16 * SPITCH + 256)
#define SMEM_BYTES  (SMEM_FLOATS * 4)
#define FRAG 512                       // elements per fragment tile: 32 lanes * 16

typedef __bf16 bf16;
typedef __attribute__((ext_vector_type(16))) __bf16 v16bf;
typedef __attribute__((ext_vector_type(8)))  float  v8f;

// ---------------------------------------------------------------------------
// CDNA5 WMMA 16x16x32 bf16 fragment maps (ISA 7.12.2):
//   A (16x32): lane = half*16 + M;  element i -> K = i + (i>=8?8:0) + 8*half
//   B (32x16): lane = half*16 + N;  K twist stored identically by the packers
//   C/D (16x16 f32): element r -> M = r + 8*half, N = lane&15
// Inverse of the K twist: c in [0,32) -> (half, i)
// ---------------------------------------------------------------------------
__device__ __forceinline__ void inv_kmap(int c, int& half, int& i) {
    half = (c >> 3) & 1;
    i    = (c & 7) + ((c >> 4) << 3);
}

__device__ __forceinline__ v8f wmma_bf16(v16bf a, v16bf b, v8f c) {
    return __builtin_amdgcn_wmma_f32_16x16x32_bf16(
        false, a, false, b, (short)0, c, false, false);
}
__device__ __forceinline__ v16bf frag_load(const bf16* __restrict__ p) {
    return *(const v16bf*)p;           // 32B aligned -> global_load_b128 x2
}

// ---------------------------------------------------------------------------
// Pack fp32 activations (B,C,HW) into B-fragment order:
//   out[b][kc=C/32][nt=HW/16][lane][16]   (K over channels, N over positions)
// ---------------------------------------------------------------------------
__global__ __launch_bounds__(256) void k_pack_x(const float* __restrict__ in,
                                                bf16* __restrict__ out) {
    int t = blockIdx.x * 256 + threadIdx.x;             // (b, c, pos) flat
    int b = t / (CCH * HWP);
    int r = t - b * (CCH * HWP);
    int c = r / HWP;
    int pos = r - c * HWP;
    int kc = c >> 5, half, i;
    inv_kmap(c & 31, half, i);
    int nt = pos >> 4;
    int lane = half * 16 + (pos & 15);
    out[((((size_t)b * (CCH / 32) + kc) * (HWP / 16) + nt) * 32 + lane) * 16 + i] =
        (bf16)in[t];
}

// Pack fp32 weights (C x C, row-major o,c) into A-fragment order:
//   out[mt=C/16][kc=C/32][lane][16]
__global__ __launch_bounds__(256) void k_pack_w(const float* __restrict__ in,
                                                bf16* __restrict__ out) {
    int t = blockIdx.x * 256 + threadIdx.x;             // (o, c) flat
    int o = t / CCH;
    int c = t - o * CCH;
    int kc = c >> 5, half, i;
    inv_kmap(c & 31, half, i);
    int lane = half * 16 + (o & 15);
    out[(((size_t)(o >> 4) * (CCH / 32) + kc) * 32 + lane) * 16 + i] = (bf16)in[t];
}

// ---------------------------------------------------------------------------
// conv1x1 GEMM on fragment-ordered operands, 32x64 output tile per wave.
//   Apack: [C/16][C/32][lane][16]    Bpack: [B][C/32][HW/16][lane][16]
//   grid = (HW/64, C/32, B), block = 32
//   Per K-step: 2 A-frags + 4 B-frags -> 8 WMMAs (0.75 b128-pairs / WMMA).
// MODE: 0 = store B-frag order, K=channels, N=positions  (k matrix)
//       1 = store A-frag order over (qpos rows, channel K) (q matrix)
//       2 = store B-frag order, K=positions, N=channels   (v matrix)
//       3 = fp32 out + bias + residual                     (final proj)
// ---------------------------------------------------------------------------
template <int MODE>
__global__ __launch_bounds__(32) void k_gemm_bf16(
    const bf16*  __restrict__ Apack,
    const bf16*  __restrict__ Bpack,
    const float* __restrict__ bias,
    bf16*        __restrict__ outb,
    float*       __restrict__ outf,
    const float* __restrict__ resid)
{
    const int lane = threadIdx.x & 31;
    const int ncol = lane & 15;
    const int half = lane >> 4;
    const int b   = blockIdx.z;
    const int mt0 = blockIdx.y * 2;      // channel tile base (o), 2 tiles
    const int nt0 = blockIdx.x * 4;      // position tile base, 4 tiles

    v8f acc[2][4] = {{v8f{}, v8f{}, v8f{}, v8f{}},
                     {v8f{}, v8f{}, v8f{}, v8f{}}};

#pragma unroll
    for (int kc = 0; kc < CCH / 32; ++kc) {
        const bf16* ap = Apack + (((size_t)mt0 * (CCH / 32) + kc) * 32 + lane) * 16;
        v16bf a0 = frag_load(ap);
        v16bf a1 = frag_load(ap + (size_t)(CCH / 32) * FRAG);
        const bf16* bp = Bpack +
            ((((size_t)b * (CCH / 32) + kc) * (HWP / 16) + nt0) * 32 + lane) * 16;
        if (kc + 1 < CCH / 32)
            __builtin_prefetch(bp + (size_t)(HWP / 16) * FRAG, 0, 1);
#pragma unroll
        for (int t = 0; t < 4; ++t) {
            v16bf bb = frag_load(bp + t * FRAG);
            acc[0][t] = wmma_bf16(a0, bb, acc[0][t]);
            acc[1][t] = wmma_bf16(a1, bb, acc[1][t]);
        }
    }

#pragma unroll
    for (int mi = 0; mi < 2; ++mi) {
#pragma unroll
        for (int t = 0; t < 4; ++t) {
#pragma unroll
            for (int r = 0; r < 8; ++r) {
                const int M = (mt0 + mi) * 16 + r + 8 * half;  // output channel
                const int N = (nt0 + t) * 16 + ncol;           // position
                float v = acc[mi][t][r] + bias[M];
                if (MODE == 3) {
                    const size_t idx = (size_t)b * CCH * HWP + (size_t)M * HWP + N;
                    outf[idx] = v + resid[idx];
                } else if (MODE == 0) {               // k: B-frags, K=ch, N=pos
                    int h2, i2; inv_kmap(M & 31, h2, i2);
                    outb[((((size_t)b * (CCH / 32) + (M >> 5)) * (HWP / 16) + (N >> 4))
                          * 32 + h2 * 16 + (N & 15)) * 16 + i2] = (bf16)v;
                } else if (MODE == 1) {               // q: A-frags over qpos x ch
                    int hq, iq; inv_kmap(M & 31, hq, iq);
                    outb[((((size_t)b * (HWP / 16) + (N >> 4)) * (CCH / 32) + (M >> 5))
                          * 32 + hq * 16 + (N & 15)) * 16 + iq] = (bf16)v;
                } else {                              // v: B-frags, K=pos, N=ch
                    int hv, iv; inv_kmap(N & 31, hv, iv);
                    outb[((((size_t)b * (HWP / 32) + (N >> 5)) * (CCH / 16) + (M >> 4))
                          * 32 + hv * 16 + (M & 15)) * 16 + iv] = (bf16)v;
                }
            }
        }
    }
}

// ---------------------------------------------------------------------------
// Attention: one workgroup (8 waves / 256 thr) per (batch, 16-query tile).
//   Phase 1: S(16x4096) = q_tile(16xC) * K(CxHW)   -> LDS fp32 (padded pitch)
//   Phase 2: row softmax in LDS
//   Phase 3: frs(16xC) = P(16x4096) * vT(HWxC)     -> frsPack (B-frag order)
// The 262.7KB score strip lives entirely in CDNA5's 320KB LDS.
// ---------------------------------------------------------------------------
__global__ __launch_bounds__(256) void k_attn(
    const bf16* __restrict__ qPack,   // [B][HW/16][C/32][lane][16]  A-frags
    const bf16* __restrict__ kPack,   // [B][C/32][HW/16][lane][16]  B-frags
    const bf16* __restrict__ vPack,   // [B][HW/32][C/16][lane][16]  B-frags
    bf16*       __restrict__ frsPack) // [B][C/32][HW/16][lane][16]  B-frags
{
    extern __shared__ float smem[];
    float* S   = smem;                 // [16][SPITCH]
    float* red = smem + 16 * SPITCH;   // [16][16]

    const int tid  = threadIdx.x;
    const int wave = tid >> 5;
    const int lane = tid & 31;
    const int ncol = lane & 15;
    const int half = lane >> 4;
    const int b     = blockIdx.y;
    const int qt    = blockIdx.x;      // query tile index
    const int qbase = qt * 16;

    // resident q A-fragments (16 q-rows x 256 channels)
    v16bf qa[CCH / 32];
    {
        const bf16* qp = qPack + (((size_t)b * (HWP / 16) + qt) * (CCH / 32) * 32
                                  + lane) * 16;
#pragma unroll
        for (int kc = 0; kc < CCH / 32; ++kc)
            qa[kc] = frag_load(qp + (size_t)kc * FRAG);
    }

    // ---- Phase 1: scores -> LDS -------------------------------------------
    const bf16* kbase = kPack + (size_t)b * (CCH / 32) * (HWP / 16) * FRAG
                              + (size_t)lane * 16;
    for (int kt = wave; kt < HWP / 16; kt += 8) {
        v8f acc = v8f{};
#pragma unroll
        for (int kc = 0; kc < CCH / 32; ++kc)
            acc = wmma_bf16(qa[kc],
                  frag_load(kbase + ((size_t)kc * (HWP / 16) + kt) * FRAG), acc);
#pragma unroll
        for (int r = 0; r < 8; ++r)
            S[(r + 8 * half) * SPITCH + kt * 16 + ncol] = acc[r];
    }
    __syncthreads();

    // ---- Phase 2: softmax over each 4096-wide row -------------------------
    {
        const int row = tid >> 4, sub = tid & 15;
        float* Srow = S + row * SPITCH;
        float mx = -3.402823466e38f;
        for (int j = sub; j < HWP; j += 16) mx = fmaxf(mx, Srow[j]);
        red[row * 16 + sub] = mx;
        __syncthreads();
        mx = -3.402823466e38f;
#pragma unroll
        for (int i = 0; i < 16; ++i) mx = fmaxf(mx, red[row * 16 + i]);
        __syncthreads();
        float sum = 0.f;
        for (int j = sub; j < HWP; j += 16) {
            float e = __expf(Srow[j] - mx);
            Srow[j] = e;
            sum += e;
        }
        red[row * 16 + sub] = sum;
        __syncthreads();
        sum = 0.f;
#pragma unroll
        for (int i = 0; i < 16; ++i) sum += red[row * 16 + i];
        const float inv = 1.0f / sum;
        for (int j = sub; j < HWP; j += 16) Srow[j] *= inv;
    }
    __syncthreads();

    // ---- Phase 3: out(16 x 256) = P(16x4096) * vT(4096x256) ---------------
    const int nb0 = wave * 32;                       // channel tile pair
    v8f acc0 = v8f{}, acc1 = v8f{};
    const float* Sr = S + ncol * SPITCH;             // P row (M = lane&15)
    const bf16* vbase = vPack + (size_t)b * (HWP / 32) * (CCH / 16) * FRAG
                              + (size_t)(nb0 >> 4) * FRAG + (size_t)lane * 16;
#pragma unroll 2
    for (int kc = 0; kc < HWP / 32; ++kc) {
        v16bf a;
        const float* sp = Sr + kc * 32 + 8 * half;   // two aligned 8-float runs
#pragma unroll
        for (int i = 0; i < 8; ++i)  a[i]     = (bf16)sp[i];
#pragma unroll
        for (int i = 0; i < 8; ++i)  a[i + 8] = (bf16)sp[i + 16];
        const bf16* bp = vbase + (size_t)kc * (CCH / 16) * FRAG;
        acc0 = wmma_bf16(a, frag_load(bp),        acc0);
        acc1 = wmma_bf16(a, frag_load(bp + FRAG), acc1);
    }
    // store into frsPack (B-frag order: K=channels, N=positions)
#pragma unroll
    for (int r = 0; r < 8; ++r) {
        const int pos = qbase + r + 8 * half;        // query position (N)
#pragma unroll
        for (int s = 0; s < 2; ++s) {
            const int ch = nb0 + s * 16 + ncol;      // channel (K)
            int hf, jf; inv_kmap(ch & 31, hf, jf);
            frsPack[((((size_t)b * (CCH / 32) + (ch >> 5)) * (HWP / 16) + qt)
                     * 32 + hf * 16 + (pos & 15)) * 16 + jf] =
                (bf16)(s == 0 ? acc0[r] : acc1[r]);
        }
    }
}

// ---------------------------------------------------------------------------
// Host-side orchestration
// ---------------------------------------------------------------------------
extern "C" void kernel_launch(void* const* d_in, const int* in_sizes, int n_in,
                              void* d_out, int out_size, void* d_ws, size_t ws_size,
                              hipStream_t stream) {
    const float* x_fcc = (const float*)d_in[0];
    const float* x_fss = (const float*)d_in[1];
    const float* w1  = (const float*)d_in[2];
    const float* b1  = (const float*)d_in[3];
    const float* w2  = (const float*)d_in[4];
    const float* b2  = (const float*)d_in[5];
    const float* w3  = (const float*)d_in[6];
    const float* b3  = (const float*)d_in[7];
    const float* wrs = (const float*)d_in[8];
    const float* brs = (const float*)d_in[9];

    const int NX = BATCH * CCH * HWP;   // 4,194,304
    const int NW = CCH * CCH;           // 65,536

    bf16* p    = (bf16*)d_ws;
    bf16* xcP  = p; p += NX;            // x_fcc, B-frag order
    bf16* xsP  = p; p += NX;            // x_fss, B-frag order
    bf16* qP   = p; p += NX;            // q, A-frag order
    bf16* kP   = p; p += NX;            // k, B-frag order (K=ch)
    bf16* vP   = p; p += NX;            // v, B-frag order (K=pos)
    bf16* frsP = p; p += NX;            // attn out, B-frag order (K=ch)
    bf16* w1P  = p; p += NW;
    bf16* w2P  = p; p += NW;
    bf16* w3P  = p; p += NW;
    bf16* wrsP = p; p += NW;

    // 1) pack activations + weights into WMMA fragment order (bf16)
    k_pack_x<<<NX / 256, 256, 0, stream>>>(x_fcc, xcP);
    k_pack_x<<<NX / 256, 256, 0, stream>>>(x_fss, xsP);
    k_pack_w<<<NW / 256, 256, 0, stream>>>(w1,  w1P);
    k_pack_w<<<NW / 256, 256, 0, stream>>>(w2,  w2P);
    k_pack_w<<<NW / 256, 256, 0, stream>>>(w3,  w3P);
    k_pack_w<<<NW / 256, 256, 0, stream>>>(wrs, wrsP);

    // 2) q/k/v projections, outputs already in attention fragment order
    dim3 ggrid(HWP / 64, CCH / 32, BATCH);
    k_gemm_bf16<1><<<ggrid, 32, 0, stream>>>(w1P, xcP, b1, qP, nullptr, nullptr);
    k_gemm_bf16<0><<<ggrid, 32, 0, stream>>>(w2P, xsP, b2, kP, nullptr, nullptr);
    k_gemm_bf16<2><<<ggrid, 32, 0, stream>>>(w3P, xsP, b3, vP, nullptr, nullptr);

    // 3) attention (score strip stays in LDS)
    (void)hipFuncSetAttribute(reinterpret_cast<const void*>(k_attn),
                              hipFuncAttributeMaxDynamicSharedMemorySize,
                              SMEM_BYTES);
    k_attn<<<dim3(HWP / 16, BATCH), 256, SMEM_BYTES, stream>>>(qP, kP, vP, frsP);

    // 4) final projection + bias + residual (fp32 out)
    k_gemm_bf16<3><<<ggrid, 32, 0, stream>>>(wrsP, frsP, brs, nullptr,
                                             (float*)d_out, x_fcc);
}
